// EmbeddingGATHead_69518340653445
// MI455X (gfx1250) — compile-verified, matrix operand order checked by reading, not run on hardware
//
#include <hip/hip_runtime.h>

// ---------------------------------------------------------------------------
// EmbeddingGATHead for MI455X (gfx1250, wave32, WMMA)
//
// Roofline: pooling reads 201MB of features, weights are 67MB f32 -> whole op
// is HBM-bound (~12us floor @ 23.3 TB/s). GEMMs are only ~6.4 GFLOP total, so
// we run them as bf16 WMMA (v_wmma_f32_16x16x32_bf16) with f32 accumulation,
// with weights pre-packed once per launch into WMMA-fragment-major bf16.
// ---------------------------------------------------------------------------

constexpr int kB = 32;      // batch
constexpr int kP = 6;       // parts per image
constexpr int kC = 2048;    // feature dim
constexpr int kHW = 128;    // H*W pooled elements
constexpr int kHeads = 4;
constexpr int kDH = 512;    // per-head dim
constexpr int kN = kB * kP; // 192 graph nodes
constexpr int kLayers = 2;

typedef __bf16 bf16;
typedef __attribute__((ext_vector_type(8)))  __bf16 v8bf;
typedef __attribute__((ext_vector_type(16))) __bf16 v16bf;
typedef __attribute__((ext_vector_type(8)))  float  v8f;

// fragments: 2 layers * 2 mats * 4 heads * 64 kblocks * 32 dblocks
constexpr int kFrags = kLayers * 2 * kHeads * (kC / 32) * (kDH / 16);

__device__ __forceinline__ float wred32(float v) {
#pragma unroll
  for (int o = 16; o > 0; o >>= 1) v += __shfl_xor(v, o, 32);
  return v;
}

// ---------------------------------------------------------------------------
// 1) Global average pool: features[B, P*C, 16, 8] -> pool[N=192, C] (f32+bf16)
//    node n = p*B + b (chunk-major, matches torch.chunk/cat in reference).
//    One wave per (n,c): 128 contiguous floats = one float4 per lane.
// ---------------------------------------------------------------------------
__global__ __launch_bounds__(256) void pool_kernel(const float* __restrict__ feat,
                                                   float* __restrict__ pool,
                                                   bf16* __restrict__ xbf) {
  int wid  = (blockIdx.x * blockDim.x + threadIdx.x) >> 5;
  int lane = threadIdx.x & 31;
  if (wid >= kN * kC) return;
  int n = wid >> 11;           // /2048
  int c = wid & (kC - 1);
  int p = n >> 5;              // n = p*32 + b
  int b = n & 31;
  size_t base = ((size_t)b * (kP * kC) + (size_t)p * kC + c) * kHW;
  float4 v = *(const float4*)(feat + base + lane * 4);
  float s = wred32(v.x + v.y + v.z + v.w) * (1.0f / 128.0f);
  if (lane == 0) {
    pool[wid] = s;
    xbf[wid]  = (bf16)s;
  }
}

// ---------------------------------------------------------------------------
// 2) Pack Wl/Wr f32 [L,H,C,DH] into WMMA B-fragment-major bf16.
//    ISA 16-bit B layout per 32x16 fragment: lane L<16 -> col d0+L,
//    k = {k0..k0+7, k0+16..k0+23}; lane>=16 -> k = {k0+8.., k0+24..}.
//    One thread = one (fragment, lane) pair -> one 32B store.
// ---------------------------------------------------------------------------
__global__ __launch_bounds__(256) void pack_w_kernel(const float* __restrict__ Wl,
                                                     const float* __restrict__ Wr,
                                                     bf16* __restrict__ packed) {
  int t    = blockIdx.x * blockDim.x + threadIdx.x;
  int lane = t & 31;
  int frag = t >> 5;
  if (frag >= kFrags) return;
  int dt    = frag & 31;
  int kb    = (frag >> 5) & 63;
  int head  = (frag >> 11) & 3;
  int mat   = (frag >> 13) & 1;
  int layer = frag >> 14;
  const float* W = (mat ? Wr : Wl) + (size_t)(layer * kHeads + head) * kC * kDH;
  int d     = dt * 16 + (lane & 15);
  int kbase = kb * 32 + (lane >> 4) * 8;
  v16bf out;
#pragma unroll
  for (int q = 0; q < 8; ++q)
    out[q] = (bf16)W[(size_t)(kbase + q) * kDH + d];
#pragma unroll
  for (int q = 0; q < 8; ++q)
    out[8 + q] = (bf16)W[(size_t)(kbase + 16 + q) * kDH + d];
  *(v16bf*)(packed + (size_t)frag * 512 + lane * 16) = out;
}

// ---------------------------------------------------------------------------
// 3) GEMM: xl/xr[n, h*512+d] = sum_k x[n,k] * W[h,k,d] + bias[h,d]
//    One wave computes a 16x16 output tile, 64 K-steps of 16x16x32 bf16 WMMA.
//    A fragment = two 16B loads of bf16 x; B fragment = one 32B packed load.
// ---------------------------------------------------------------------------
__global__ __launch_bounds__(256) void gemm_kernel(const bf16* __restrict__ xbf,
                                                   const bf16* __restrict__ packed,
                                                   const float* __restrict__ bl,
                                                   const float* __restrict__ br,
                                                   int layer,
                                                   float* __restrict__ outL,
                                                   float* __restrict__ outR) {
  int wid  = (blockIdx.x * blockDim.x + threadIdx.x) >> 5;
  int lane = threadIdx.x & 31;
  if (wid >= 2 * (kN / 16) * (kC / 16)) return;   // 3072 tiles
  int mat   = wid / 1536;
  int rem   = wid % 1536;
  int mtile = rem >> 7;          // 12 row tiles
  int ct    = rem & 127;         // 128 col tiles
  int head  = ct >> 5;
  int dt    = ct & 31;

  int half = lane >> 4;
  int lr   = lane & 15;

  const bf16* ap = xbf + (size_t)(mtile * 16 + lr) * kC + half * 8;
  size_t fragbase = ((size_t)(((layer * 2 + mat) * kHeads + head) * 64) * 32 + dt) * 512;
  const bf16* bp = packed + fragbase + lane * 16;

  v8f acc = {0.f, 0.f, 0.f, 0.f, 0.f, 0.f, 0.f, 0.f};
#pragma unroll 4
  for (int kb = 0; kb < 64; ++kb) {
    v8bf lo = *(const v8bf*)(ap);
    v8bf hi = *(const v8bf*)(ap + 16);
    v16bf a;
#pragma unroll
    for (int q = 0; q < 8; ++q) { a[q] = lo[q]; a[q + 8] = hi[q]; }
    v16bf bfr = *(const v16bf*)(bp);
    __builtin_prefetch(bp + 32 * 512, 0, 1);   // next K-step B fragment
    acc = __builtin_amdgcn_wmma_f32_16x16x32_bf16(false, a, false, bfr,
                                                  (short)0, acc, false, false);
    ap += 32;
    bp += 32 * 512;
  }

  float bv = ((mat ? br : bl) + (size_t)(layer * kHeads + head) * kDH)[dt * 16 + lr];
  float* out = mat ? outR : outL;
  int col  = head * kDH + dt * 16 + lr;
  int row0 = mtile * 16 + half * 8;   // C/D layout: vgpr r -> M=r (lanes<16) / M=8+r
#pragma unroll
  for (int r = 0; r < 8; ++r)
    out[(size_t)(row0 + r) * kC + col] = acc[r] + bv;
}

// ---------------------------------------------------------------------------
// 4) GATv2 attention over 6-node clusters. One wave per (node i, head h);
//    lane owns 16 of the 512 head dims. Masked softmax over the 6 cluster
//    sources (+ guaranteed self loop), then alpha-weighted sum of xl.
// ---------------------------------------------------------------------------
__global__ __launch_bounds__(256) void attn_kernel(const float* __restrict__ xl,
                                                   const float* __restrict__ xr,
                                                   const float* __restrict__ att,
                                                   const float* __restrict__ gbias,
                                                   const int* __restrict__ nps,
                                                   int layer,
                                                   float* __restrict__ outf,
                                                   bf16* __restrict__ outbf) {
  int wid  = (blockIdx.x * blockDim.x + threadIdx.x) >> 5;
  int lane = threadIdx.x & 31;
  if (wid >= kN * kHeads) return;
  int i = wid >> 2;
  int h = wid & 3;
  int d0 = h * kDH + lane * 16;

  float av[16], rv[16];
  const float* ap = att + (size_t)(layer * kHeads + h) * kDH + lane * 16;
  const float* rp = xr + (size_t)i * kC + d0;
#pragma unroll
  for (int q = 0; q < 4; ++q) {
    float4 fa = *(const float4*)(ap + q * 4);
    float4 fr = *(const float4*)(rp + q * 4);
    av[q*4+0]=fa.x; av[q*4+1]=fa.y; av[q*4+2]=fa.z; av[q*4+3]=fa.w;
    rv[q*4+0]=fr.x; rv[q*4+1]=fr.y; rv[q*4+2]=fr.z; rv[q*4+3]=fr.w;
  }

  int inst = i / kP;
  int c0   = inst * kP;
  int np   = nps[inst];
  bool vi  = (i - c0) < np;

  float xlv[kP][16];
  float s[kP];
#pragma unroll
  for (int j = 0; j < kP; ++j) {
    int jn = c0 + j;
    const float* xp = xl + (size_t)jn * kC + d0;
#pragma unroll
    for (int q = 0; q < 4; ++q) {
      float4 f = *(const float4*)(xp + q * 4);
      xlv[j][q*4+0]=f.x; xlv[j][q*4+1]=f.y; xlv[j][q*4+2]=f.z; xlv[j][q*4+3]=f.w;
    }
    float part = 0.f;
#pragma unroll
    for (int t = 0; t < 16; ++t) {
      float z = rv[t] + xlv[j][t];
      z = z > 0.f ? z : 0.2f * z;          // leaky_relu(., 0.2)
      part += z * av[t];
    }
    part = wred32(part);
    bool vj = j < np;
    s[j] = ((vi && vj) || (jn == i)) ? part : -1e30f;
  }

  float m = s[0];
#pragma unroll
  for (int j = 1; j < kP; ++j) m = fmaxf(m, s[j]);
  float e[kP], sum = 0.f;
#pragma unroll
  for (int j = 0; j < kP; ++j) { e[j] = __expf(s[j] - m); sum += e[j]; }
  float inv = 1.f / sum;

  float ov[16];
#pragma unroll
  for (int t = 0; t < 16; ++t) {
    float v = 0.f;
#pragma unroll
    for (int j = 0; j < kP; ++j) v += e[j] * xlv[j][t];
    v = v * inv + gbias[(size_t)layer * kC + d0 + t];
    if (layer == 0) v = v > 0.f ? v : expm1f(v);   // ELU between layers
    ov[t] = v;
  }
  if (outf) {
    float* op = outf + (size_t)i * kC + d0;
#pragma unroll
    for (int q = 0; q < 4; ++q)
      *(float4*)(op + q * 4) = make_float4(ov[q*4+0], ov[q*4+1], ov[q*4+2], ov[q*4+3]);
  }
  if (outbf) {
    v16bf ob;
#pragma unroll
    for (int t = 0; t < 16; ++t) ob[t] = (bf16)ov[t];
    *(v16bf*)(outbf + (size_t)i * kC + d0) = ob;
  }
}

// ---------------------------------------------------------------------------
// 5) Residual + mean over 6 parts + frozen BN -> out[B, C]
// ---------------------------------------------------------------------------
__global__ __launch_bounds__(256) void final_kernel(const float* __restrict__ xfin,
                                                    const float* __restrict__ pool,
                                                    const float* __restrict__ gamma,
                                                    const float* __restrict__ mean,
                                                    const float* __restrict__ var,
                                                    float* __restrict__ out) {
  int t = blockIdx.x * blockDim.x + threadIdx.x;
  if (t >= kB * kC) return;
  int b = t >> 11;
  int c = t & (kC - 1);
  float s = 0.f;
#pragma unroll
  for (int p = 0; p < kP; ++p) {
    size_t idx = (size_t)(p * kB + b) * kC + c;
    s += xfin[idx] + pool[idx];
  }
  float g = s * (1.0f / 6.0f);
  out[t] = gamma[c] * (g - mean[c]) * rsqrtf(var[c] + 1e-5f);
}

// ---------------------------------------------------------------------------
extern "C" void kernel_launch(void* const* d_in, const int* in_sizes, int n_in,
                              void* d_out, int out_size, void* d_ws, size_t ws_size,
                              hipStream_t stream) {
  const float* feat  = (const float*)d_in[0];
  const int*   nps   = (const int*)d_in[1];
  const float* Wl    = (const float*)d_in[2];
  const float* bl    = (const float*)d_in[3];
  const float* Wr    = (const float*)d_in[4];
  const float* br    = (const float*)d_in[5];
  const float* att   = (const float*)d_in[6];
  const float* gbias = (const float*)d_in[7];
  const float* gamma = (const float*)d_in[8];
  const float* bmean = (const float*)d_in[9];
  const float* bvar  = (const float*)d_in[10];
  (void)in_sizes; (void)n_in; (void)out_size; (void)ws_size;

  char* ws = (char*)d_ws;
  size_t off = 0;
  auto alloc = [&](size_t bytes) -> char* {
    char* p = ws + off;
    off += (bytes + 255) & ~(size_t)255;
    return p;
  };
  float* pool   = (float*)alloc((size_t)kN * kC * 4);
  bf16*  xbf0   = (bf16*) alloc((size_t)kN * kC * 2);
  bf16*  xbf1   = (bf16*) alloc((size_t)kN * kC * 2);
  bf16*  packed = (bf16*) alloc((size_t)kFrags * 512 * 2);   // ~33.5 MB
  float* xlb    = (float*)alloc((size_t)kN * kC * 4);
  float* xrb    = (float*)alloc((size_t)kN * kC * 4);
  float* xfin   = (float*)alloc((size_t)kN * kC * 4);

  // 1) pool (one wave per (n,c)): 393216 waves -> 49152 blocks
  pool_kernel<<<(kN * kC) / 8, 256, 0, stream>>>(feat, pool, xbf0);

  // 2) pack all weights to bf16 fragments: kFrags*32 threads
  pack_w_kernel<<<(kFrags * 32) / 256, 256, 0, stream>>>(Wl, Wr, packed);

  // 3) layer 0: GEMM (3072 tiles / 8 waves per block) + attention
  gemm_kernel<<<384, 256, 0, stream>>>(xbf0, packed, bl, br, 0, xlb, xrb);
  attn_kernel<<<(kN * kHeads) / 8, 256, 0, stream>>>(xlb, xrb, att, gbias, nps, 0,
                                                     nullptr, xbf1);

  // 4) layer 1
  gemm_kernel<<<384, 256, 0, stream>>>(xbf1, packed, bl, br, 1, xlb, xrb);
  attn_kernel<<<(kN * kHeads) / 8, 256, 0, stream>>>(xlb, xrb, att, gbias, nps, 1,
                                                     xfin, nullptr);

  // 5) residual + part-mean + BN
  final_kernel<<<(kB * kC) / 256, 256, 0, stream>>>(xfin, pool, gamma, bmean, bvar,
                                                    (float*)d_out);
}
